// Sparse_ConvFuser_75033078661475
// MI455X (gfx1250) — matrix operand order repaired and checked by reading, not compiled, not run on hardware
//
#include <hip/hip_runtime.h>
#include <hip/hip_bf16.h>

typedef __attribute__((ext_vector_type(16))) _Float16 v16h;
typedef __attribute__((ext_vector_type(8)))  _Float16 v8h;
typedef __attribute__((ext_vector_type(8)))  float    v8f;

#define HXW   32400      // 180*180
#define WIMG  180
#define CLID  128
#define CCAM  128
#define CHID  170
#define COUT  256
#define AP    264        // padded row (halves) for LDS tiles: 264*2B = 528B = 33*16B

// ---- workspace layout (bytes) ----
#define IMG_OFF   0                          // f16 [32400][128]  (batch-0 plane, transposed)
#define IMG_BYTES (HXW * 128 * 2)            // 8,294,400
#define MM_OFF    (IMG_OFF + IMG_BYTES)      // u32 [4 kinds][4 batches][128]
#define MM_BYTES  (4 * 4 * 128 * 4)
#define ATT_OFF   (MM_OFF + MM_BYTES)        // f32 [2][4][128]  (att_l, att_c)
#define ATT_BYTES (2 * 4 * 128 * 4)
#define WF_OFF    (ATT_OFF + ATT_BYTES)      // f16 [2][8 kb][16 nb][32 lane][16 h]
#define WF_BYTES  (2 * 8 * 16 * 32 * 16 * 2)

// ---- order-preserving float<->uint mapping for atomic min/max ----
__device__ __forceinline__ unsigned f2ord(float f) {
    unsigned u = __float_as_uint(f);
    return (u & 0x80000000u) ? ~u : (u | 0x80000000u);
}
__device__ __forceinline__ float ord2f(unsigned o) {
    unsigned u = (o & 0x80000000u) ? (o & 0x7FFFFFFFu) : ~o;
    return __uint_as_float(u);
}

// ================= K0: init min/max accumulators =================
__global__ void k_init_mm(unsigned* mm) {
    int idx = blockIdx.x * 256 + threadIdx.x;
    if (idx < 2048) {
        int kind = idx >> 9;                 // 0=lmin 1=lmax 2=cmin 3=cmax
        mm[idx] = (kind == 0 || kind == 2) ? 0xFFFFFFFFu : 0u;
    }
}

// ================= K1: transpose img_bev[0] -> f16 [HW][128] =================
__global__ void k_img2f16(const float* __restrict__ img, _Float16* __restrict__ wsImg) {
    __shared__ float tile[32][65];
    int p0 = blockIdx.x * 64;
    int c0 = blockIdx.y * 32;
    int tx = threadIdx.x & 63;
    int ty = threadIdx.x >> 6;               // 0..3
    for (int i = 0; i < 8; ++i) {
        int ch = ty * 8 + i;                 // 0..31
        int p  = p0 + tx;
        tile[ch][tx] = (p < HXW) ? img[(size_t)(c0 + ch) * HXW + p] : 0.f;
    }
    __syncthreads();
    int cc = threadIdx.x & 31;
    int pr = threadIdx.x >> 5;               // 0..7
    for (int j = 0; j < 8; ++j) {
        int p  = pr + j * 8;
        int gp = p0 + p;
        if (gp < HXW)
            wsImg[(size_t)gp * 128 + c0 + cc] = (_Float16)tile[cc][p];
    }
}

// ================= K2: segmented min/max =================
__device__ __forceinline__ void flushmm(unsigned* mm, int b, int c,
                                        float lmin, float lmax, float cmin, float cmax) {
    if (b < 0) return;
    atomicMin(&mm[0 * 512 + b * 128 + c], f2ord(lmin));
    atomicMax(&mm[1 * 512 + b * 128 + c], f2ord(lmax));
    atomicMin(&mm[2 * 512 + b * 128 + c], f2ord(cmin));
    atomicMax(&mm[3 * 512 + b * 128 + c], f2ord(cmax));
}

__global__ void k_minmax(const float* __restrict__ lidar, const _Float16* __restrict__ wsImg,
                         const int* __restrict__ bidx, const int* __restrict__ yidx,
                         const int* __restrict__ xidx, unsigned* __restrict__ mm, int N) {
    int c  = threadIdx.x;                    // 0..127 (channel)
    int v0 = blockIdx.x * 256;
    int cur = -1;
    float lmin = 3.4e38f, lmax = -3.4e38f, cmin = 3.4e38f, cmax = -3.4e38f;
    for (int r = 0; r < 256; ++r) {
        int v = v0 + r;
        if (v >= N) break;
        int b = bidx[v];
        if (b != cur) {
            flushmm(mm, cur, c, lmin, lmax, cmin, cmax);
            cur = b;
            lmin = 3.4e38f; lmax = -3.4e38f; cmin = 3.4e38f; cmax = -3.4e38f;
        }
        float lf = lidar[(size_t)v * CLID + c];
        int row = b * HXW + yidx[v] * WIMG + xidx[v];
        row = row < 0 ? 0 : row;
        row = row > HXW - 2 ? HXW - 2 : row; // reference clamp: always batch-0 plane
        float cf = (float)wsImg[(size_t)row * 128 + c];
        lmin = fminf(lmin, lf); lmax = fmaxf(lmax, lf);
        cmin = fminf(cmin, cf); cmax = fmaxf(cmax, cf);
    }
    flushmm(mm, cur, c, lmin, lmax, cmin, cmax);
}

// ================= K3: swizzle Wf1/Wf2 into WMMA B-fragment layout =================
// B layout (16x16x32 f16): lane<16: n=lane, K={0..7,16..23} ; lane>=16: n=lane-16, K={8..15,24..31}
__global__ void k_swizzleW(const float* __restrict__ Wf1, const float* __restrict__ Wf2,
                           _Float16* __restrict__ wfrag) {
    int tid = blockIdx.x * 256 + threadIdx.x;        // < 131072
    int h    = tid & 15;
    int lane = (tid >> 4) & 31;
    int nb   = (tid >> 9) & 15;
    int kb   = (tid >> 13) & 7;
    int w    = tid >> 16;
    int n = nb * 16 + (lane & 15);
    int k = kb * 32 + (h & 7) + ((lane >> 4) << 3) + ((h >> 3) << 4);
    const float* W = w ? Wf2 : Wf1;                  // stored [in(256)][out(256)]
    wfrag[tid] = (_Float16)W[(size_t)k * 256 + n];
}

// ================= K4: attention MLP (B=4, tiny) =================
__global__ void k_mlp(const unsigned* __restrict__ mm,
                      const float* __restrict__ Wl1, const float* __restrict__ Wl2,
                      const float* __restrict__ Wc1, const float* __restrict__ Wc2,
                      float* __restrict__ att) {
    __shared__ float cat[4][512];
    __shared__ float hid[4][CHID];
    int t = threadIdx.x;
    int branch = blockIdx.x;                          // 0 = lidar, 1 = camera
    for (int idx = t; idx < 2048; idx += 256) {
        int b = idx >> 9, j = idx & 511;
        int kind = j >> 7, ch = j & 127;
        cat[b][j] = ord2f(mm[kind * 512 + b * 128 + ch]);
    }
    __syncthreads();
    const float* W1 = branch ? Wc1 : Wl1;             // [512][170]
    for (int idx = t; idx < 4 * CHID; idx += 256) {
        int b = idx / CHID, o = idx % CHID;
        float s = 0.f;
        for (int k = 0; k < 512; ++k) s += cat[b][k] * W1[(size_t)k * CHID + o];
        hid[b][o] = fmaxf(s, 0.f);
    }
    __syncthreads();
    const float* W2 = branch ? Wc2 : Wl2;             // [170][128]
    for (int idx = t; idx < 512; idx += 256) {
        int b = idx >> 7, o = idx & 127;
        float s = 0.f;
        for (int k = 0; k < CHID; ++k) s += hid[b][k] * W2[(size_t)k * 128 + o];
        s = fmaxf(s, 0.f);
        att[branch * 512 + b * 128 + o] = 1.f / (1.f + expf(-s));
    }
}

// ================= K5: main fused gate + double GEMM (WMMA, 4x4 register tile) ===
// Each wave owns 4 M-subtiles (64 voxels). Per 64-col N-block it holds 16
// accumulators and per K-step loads 4 A frags (LDS) + 4 B frags (global) ->
// 16 WMMAs, i.e. 0.25KB A + 0.25KB B per wmma instead of 1KB B per wmma.
union V16 { v16h v; v8h h[2]; };

__global__ __launch_bounds__(64) void k_main(
        const float* __restrict__ lidar, const _Float16* __restrict__ wsImg,
        const int* __restrict__ bidx, const int* __restrict__ yidx,
        const int* __restrict__ xidx, const float* __restrict__ attL,
        const float* __restrict__ attC, const _Float16* __restrict__ wfrag,
        float* __restrict__ out, int N) {
    __shared__ __align__(16) _Float16 Atile[2][4][16][AP];
    __shared__ __align__(16) _Float16 Htile[2][4][16][AP];
    int wave = threadIdx.x >> 5;
    int lane = threadIdx.x & 31;
    int tilebase = (blockIdx.x * 2 + wave) * 64;      // 64 voxels per wave

    // ---- build gated fused tiles A = [gated_lidar | gated_cam] as f16 ----
    for (int v = 0; v < 64; ++v) {
        int voxel = tilebase + v;
        bool ok = voxel < N;
        int b = ok ? bidx[voxel] : 0;
        int row = 0;
        if (ok) {
            row = b * HXW + yidx[voxel] * WIMG + xidx[voxel];
            row = row < 0 ? 0 : row;
            row = row > HXW - 2 ? HXW - 2 : row;
        }
        for (int it = 0; it < 8; ++it) {
            int c = it * 32 + lane;
            float val = 0.f;
            if (ok) {
                if (c < 128) val = lidar[(size_t)voxel * CLID + c] * attL[b * 128 + c];
                else         val = (float)wsImg[(size_t)row * 128 + (c - 128)] * attC[b * 128 + (c - 128)];
            }
            Atile[wave][v >> 4][v & 15][c] = (_Float16)val;
        }
    }
    __syncthreads();

    int m  = lane & 15;
    int hi = lane >> 4;                       // A: lanes>=16 hold K {8..15, 24..31}
    int mb = hi * 8;

    // ---- GEMM1: H = relu(A @ Wf1) ----
    for (int nblk = 0; nblk < 4; ++nblk) {
        v8f acc[16];
        #pragma unroll
        for (int i = 0; i < 16; ++i) acc[i] = {};
        #pragma unroll
        for (int kb = 0; kb < 8; ++kb) {
            int k0 = kb * 32 + (hi << 3);
            V16 a[4];
            #pragma unroll
            for (int mt = 0; mt < 4; ++mt) {
                a[mt].h[0] = *(const v8h*)&Atile[wave][mt][m][k0];
                a[mt].h[1] = *(const v8h*)&Atile[wave][mt][m][k0 + 16];
            }
            v16h bf[4];
            #pragma unroll
            for (int nn = 0; nn < 4; ++nn) {
                int nb = nblk * 4 + nn;
                bf[nn] = *(const v16h*)(wfrag + (size_t)((kb * 16 + nb) * 32 + lane) * 16);
            }
            #pragma unroll
            for (int mt = 0; mt < 4; ++mt)
                #pragma unroll
                for (int nn = 0; nn < 4; ++nn)
                    acc[mt * 4 + nn] = __builtin_amdgcn_wmma_f32_16x16x32_f16(
                        false, a[mt].v, false, bf[nn], (short)0, acc[mt * 4 + nn], false, false);
        }
        #pragma unroll
        for (int mt = 0; mt < 4; ++mt) {
            #pragma unroll
            for (int nn = 0; nn < 4; ++nn) {
                int n = (nblk * 4 + nn) * 16 + m;
                #pragma unroll
                for (int r = 0; r < 8; ++r) {
                    float x = acc[mt * 4 + nn][r];
                    Htile[wave][mt][mb + r][n] = (_Float16)(x > 0.f ? x : 0.f);
                }
            }
        }
    }
    __syncthreads();

    // ---- GEMM2: out = relu(H @ Wf2) ----
    for (int nblk = 0; nblk < 4; ++nblk) {
        v8f acc[16];
        #pragma unroll
        for (int i = 0; i < 16; ++i) acc[i] = {};
        #pragma unroll
        for (int kb = 0; kb < 8; ++kb) {
            int k0 = kb * 32 + (hi << 3);
            V16 a[4];
            #pragma unroll
            for (int mt = 0; mt < 4; ++mt) {
                a[mt].h[0] = *(const v8h*)&Htile[wave][mt][m][k0];
                a[mt].h[1] = *(const v8h*)&Htile[wave][mt][m][k0 + 16];
            }
            v16h bf[4];
            #pragma unroll
            for (int nn = 0; nn < 4; ++nn) {
                int nb = nblk * 4 + nn;
                bf[nn] = *(const v16h*)(wfrag + (size_t)((128 + kb * 16 + nb) * 32 + lane) * 16);
            }
            #pragma unroll
            for (int mt = 0; mt < 4; ++mt)
                #pragma unroll
                for (int nn = 0; nn < 4; ++nn)
                    acc[mt * 4 + nn] = __builtin_amdgcn_wmma_f32_16x16x32_f16(
                        false, a[mt].v, false, bf[nn], (short)0, acc[mt * 4 + nn], false, false);
        }
        #pragma unroll
        for (int mt = 0; mt < 4; ++mt) {
            #pragma unroll
            for (int nn = 0; nn < 4; ++nn) {
                int n = (nblk * 4 + nn) * 16 + m;
                #pragma unroll
                for (int r = 0; r < 8; ++r) {
                    int vrow = tilebase + mt * 16 + mb + r;
                    float x = acc[mt * 4 + nn][r];
                    if (vrow < N) out[(size_t)vrow * COUT + n] = (x > 0.f ? x : 0.f);
                }
            }
        }
    }
}

// ================= host launcher =================
extern "C" void kernel_launch(void* const* d_in, const int* in_sizes, int n_in,
                              void* d_out, int out_size, void* d_ws, size_t ws_size,
                              hipStream_t stream) {
    const float* lidar = (const float*)d_in[0];
    const float* img   = (const float*)d_in[1];
    const int*   bidx  = (const int*)d_in[2];
    const int*   yidx  = (const int*)d_in[3];
    const int*   xidx  = (const int*)d_in[4];
    const float* Wl1   = (const float*)d_in[5];
    const float* Wl2   = (const float*)d_in[6];
    const float* Wc1   = (const float*)d_in[7];
    const float* Wc2   = (const float*)d_in[8];
    const float* Wf1   = (const float*)d_in[9];
    const float* Wf2   = (const float*)d_in[10];
    float* out = (float*)d_out;
    int N = in_sizes[2];

    char* wsb = (char*)d_ws;
    _Float16* wsImg = (_Float16*)(wsb + IMG_OFF);
    unsigned* mm    = (unsigned*)(wsb + MM_OFF);
    float*    att   = (float*)(wsb + ATT_OFF);
    _Float16* wfrag = (_Float16*)(wsb + WF_OFF);

    k_init_mm<<<dim3(8), dim3(256), 0, stream>>>(mm);
    k_img2f16<<<dim3((HXW + 63) / 64, 4), dim3(256), 0, stream>>>(img, wsImg);
    k_minmax<<<dim3((N + 255) / 256), dim3(128), 0, stream>>>(lidar, wsImg, bidx, yidx, xidx, mm, N);
    k_swizzleW<<<dim3(512), dim3(256), 0, stream>>>(Wf1, Wf2, wfrag);
    k_mlp<<<dim3(2), dim3(256), 0, stream>>>(mm, Wl1, Wl2, Wc1, Wc2, att);
    k_main<<<dim3((N + 127) / 128), dim3(64), 0, stream>>>(lidar, wsImg, bidx, yidx, xidx,
                                                           att, att + 512, wfrag, out, N);
}
